// Decoder_67740224192742
// MI455X (gfx1250) — matrix operand order repaired and checked by reading, not compiled, not run on hardware
//
#include <hip/hip_runtime.h>

namespace {

constexpr int EE = 256;   // E
constexpr int HH = 512;   // H
constexpr int LL = 1024;  // L
constexpr int BB = 64;    // B
constexpr int TT = 20;    // T

typedef _Float16 f16;
typedef f16   v16h __attribute__((ext_vector_type(16)));
typedef float v8f  __attribute__((ext_vector_type(8)));

__device__ __forceinline__ float sigf(float x) { return 1.0f / (1.0f + expf(-x)); }

// ---------------------------------------------------------------------------
// Generic WMMA GEMM: C[m,n] (+)= sum_k A[m,k]*B[k,n] (+ bias), f16 in, f32 acc.
//   BT=true : B stored as (n,k) row-major, ldb = row stride (K contiguous).
//   BT=false: B stored as (k,n) row-major, ldb = row stride (N contiguous).
// Wave computes one 16x16 tile; block = 8 waves -> 16(M) x 128(N).
// Grid: x = N/128, y = M/16, z = batch (sA/sB/sC batch strides in elements).
// M,N assumed multiples of 16/128 as launched; K multiple of 32.
// ---------------------------------------------------------------------------
template <bool BT>
__global__ void wmma_gemm_kernel(const float* __restrict__ A, int lda, long sA,
                                 const float* __restrict__ Bm, int ldb, long sB,
                                 float* __restrict__ C, int ldc, long sC,
                                 const float* __restrict__ bias, int biasPerRow,
                                 int accumulate, int K) {
  const int lane = threadIdx.x & 31;
  const int wave = threadIdx.x >> 5;
  const int n0 = blockIdx.x * 128 + wave * 16;
  const int m0 = blockIdx.y * 16;
  const long bz = blockIdx.z;
  A  += bz * sA;
  Bm += bz * sB;
  C  += bz * sC;
  const int half = lane >> 4;   // which half-wave (selects K groups)
  const int lc   = lane & 15;
  const int m = m0 + lc;        // A-fragment row owned by this lane
  const int n = n0 + lc;        // B-fragment col / C col owned by this lane

  v8f acc;
#pragma unroll
  for (int r = 0; r < 8; ++r) {
    float v = 0.0f;
    if (accumulate) v = C[(long)(m0 + r + half * 8) * ldc + n];
    if (bias) v += biasPerRow ? bias[m0 + r + half * 8] : bias[n];
    acc[r] = v;
  }

#pragma unroll 2
  for (int kk = 0; kk < K; kk += 32) {
    v16h a, b;
    // 16-bit A 16x32 layout: lanes 0-15 rows M, halves hold K 0..15
    // and 16..31 respectively (contiguous, matching B).
    const float* ap = A + (long)m * lda + kk + half * 16;
#pragma unroll
    for (int i = 0; i < 16; ++i) a[i] = (f16)ap[i];
    if (BT) {
      // B 32x16: lane holds col n, halves hold K 0..15 / 16..31 (contiguous).
      const float* bp = Bm + (long)n * ldb + kk + half * 16;
#pragma unroll
      for (int i = 0; i < 16; ++i) b[i] = (f16)bp[i];
    } else {
      const float* bp = Bm + (long)(kk + half * 16) * ldb + n;
#pragma unroll
      for (int i = 0; i < 16; ++i) b[i] = (f16)bp[(long)i * ldb];
    }
    acc = __builtin_amdgcn_wmma_f32_16x16x32_f16(false, a, false, b, (short)0,
                                                 acc, false, false);
  }

#pragma unroll
  for (int r = 0; r < 8; ++r)
    C[(long)(m0 + r + half * 8) * ldc + n] = acc[r];
}

// ---------------------------------------------------------------------------
// LSTM pointwise: gates (B,4H) order [i,f,c,o]; updates c,h in place.
// ---------------------------------------------------------------------------
__global__ void lstm_kernel(const float* __restrict__ gates, float* __restrict__ c,
                            float* __restrict__ h) {
  int i = blockIdx.x * blockDim.x + threadIdx.x;  // i < BB*HH
  int b = i / HH, hh = i % HH;
  const float* g = gates + (long)b * 4 * HH;
  float ig = sigf(g[hh]);
  float fg = sigf(g[HH + hh]);
  float cg = tanhf(g[2 * HH + hh]);
  float og = sigf(g[3 * HH + hh]);
  float cy = fg * c[i] + ig * cg;
  c[i] = cy;
  h[i] = og * tanhf(cy);
}

// ---------------------------------------------------------------------------
// Attention scoring + masked softmax + argmax. One block (1024 thr) per b.
// logits[b,l] = sum_h v[h]*tanh(q[b,h] + e[b,h,l]); e layout (B,H,L).
// Writes probs, selection (as float), selection index, and sets mask[b,sel].
// ---------------------------------------------------------------------------
__global__ void attn_kernel(const float* __restrict__ e, const float* __restrict__ q,
                            const float* __restrict__ v, unsigned char* mask,
                            float* __restrict__ probs, float* __restrict__ sel_out,
                            int* __restrict__ idx_out) {
  const int b = blockIdx.x;
  const int tid = threadIdx.x;  // 0..1023 == l
  __shared__ float sq[HH];
  __shared__ float sv[HH];
  __shared__ float sr[LL];
  __shared__ int si[LL];
  if (tid < HH) {
    sq[tid] = q[b * HH + tid];
    sv[tid] = v[tid];
  }
  __syncthreads();

  const float* er = e + (size_t)b * HH * LL + tid;
  float acc = 0.0f;
  for (int h0 = 0; h0 < HH; h0 += 8) {
    // Speculative prefetch 64 rows (256KB) ahead of the 4KB-strided stream;
    // OOB prefetches are silently dropped.
    __builtin_prefetch(er + (size_t)(h0 + 64) * LL, 0, 0);
#pragma unroll
    for (int j = 0; j < 8; ++j) {
      const int hh = h0 + j;
      acc += sv[hh] * tanhf(sq[hh] + er[(size_t)hh * LL]);
    }
  }

  const bool msk = mask[b * LL + tid] != 0;
  float logit = msk ? -3.0e38f : acc;

  // max + first-index argmax
  sr[tid] = logit;
  si[tid] = tid;
  __syncthreads();
  for (int s = 512; s > 0; s >>= 1) {
    if (tid < s) {
      float v2 = sr[tid + s];
      int i2 = si[tid + s];
      if (v2 > sr[tid] || (v2 == sr[tid] && i2 < si[tid])) {
        sr[tid] = v2;
        si[tid] = i2;
      }
    }
    __syncthreads();
  }
  const float mx = sr[0];
  const int mi = si[0];
  __syncthreads();

  float ex = msk ? 0.0f : expf(logit - mx);
  sr[tid] = ex;
  __syncthreads();
  for (int s = 512; s > 0; s >>= 1) {
    if (tid < s) sr[tid] += sr[tid + s];
    __syncthreads();
  }
  probs[(size_t)b * LL + tid] = ex / sr[0];

  if (tid == 0) {
    sel_out[b] = (float)mi;
    idx_out[b] = mi;
    mask[b * LL + mi] = 1;  // applies to subsequent steps only
  }
}

// ---------------------------------------------------------------------------
// g[b,h] = sum_l e[b,h,l] * probs[b,l].  Grid (B, H/8), block 256 (8 waves).
// probs row (4KB) is staged into LDS with an async global->LDS copy so the
// 8 waves of the block read it from LDS instead of 8x from L2.
// ---------------------------------------------------------------------------
__global__ void wsum_kernel(const float* __restrict__ e,
                            const float* __restrict__ probs,
                            float* __restrict__ g) {
  const int b = blockIdx.x;
  const int tid = threadIdx.x;  // 0..255
  __shared__ float sprobs[LL];

  // 256 lanes x 16B = 4KB async copy into LDS. Low 32 bits of a flat
  // __shared__ pointer are the LDS byte offset (aperture layout).
  {
    const float* gsrc = probs + (size_t)b * LL + tid * 4;
    unsigned lds_off = (unsigned)(uintptr_t)(&sprobs[0]) + tid * 16;
    asm volatile("global_load_async_to_lds_b128 %0, %1, off"
                 :: "v"(lds_off), "v"(gsrc) : "memory");
    asm volatile("s_wait_asynccnt 0x0" ::: "memory");
  }
  __syncthreads();

  const int hh = blockIdx.y * 8 + (tid >> 5);
  const int lane = tid & 31;
  const float* er = e + ((size_t)b * HH + hh) * LL;
  float acc = 0.0f;
#pragma unroll 4
  for (int l = lane; l < LL; l += 32) acc += er[l] * sprobs[l];
#pragma unroll
  for (int off = 16; off > 0; off >>= 1) acc += __shfl_down(acc, off, 32);
  if (lane == 0) g[b * HH + hh] = acc;
}

// ---------------------------------------------------------------------------
// cls[b,c] = sum_k gs[b,k]*W[c,0,k] + ge[b,k]*W[c,1,k] + bc[c]
// plus gather: x[b,:] = emb[idx[b], b, :].  One block (256 thr) per b.
// ---------------------------------------------------------------------------
__global__ void cls_gather_kernel(const float* __restrict__ gs,
                                  const float* __restrict__ ge,
                                  const float* __restrict__ Wc,
                                  const float* __restrict__ bc,
                                  const float* __restrict__ emb,
                                  const int* __restrict__ idxh,
                                  float* __restrict__ cls_out,
                                  float* __restrict__ xbuf) {
  const int b = blockIdx.x;
  const int tid = threadIdx.x;  // 0..255
  __shared__ float s0[256];
  __shared__ float s1[256];
  float p0 = 0.0f, p1 = 0.0f;
  for (int k = tid; k < HH; k += 256) {
    float a = gs[b * HH + k], e2 = ge[b * HH + k];
    p0 += a * Wc[k]          + e2 * Wc[HH + k];
    p1 += a * Wc[2 * HH + k] + e2 * Wc[3 * HH + k];
  }
  s0[tid] = p0;
  s1[tid] = p1;
  __syncthreads();
  for (int s = 128; s > 0; s >>= 1) {
    if (tid < s) { s0[tid] += s0[tid + s]; s1[tid] += s1[tid + s]; }
    __syncthreads();
  }
  if (tid == 0) {
    cls_out[b * 2 + 0] = s0[0] + bc[0];
    cls_out[b * 2 + 1] = s1[0] + bc[1];
  }
  const int idx = idxh[b];
  xbuf[b * EE + tid] = emb[((size_t)idx * BB + b) * EE + tid];
}

__global__ void init_kernel(const float* __restrict__ dec, const float* __restrict__ h0,
                            const float* __restrict__ c0, float* __restrict__ x,
                            float* __restrict__ h, float* __restrict__ c,
                            unsigned char* mh, unsigned char* mt) {
  int i = blockIdx.x * blockDim.x + threadIdx.x;  // grid sized to BB*LL
  if (i < BB * EE) x[i] = dec[i];
  if (i < BB * HH) { h[i] = h0[i]; c[i] = c0[i]; }
  mh[i] = 0;
  mt[i] = 0;
}

__global__ void fin_kernel(const float* __restrict__ h, const float* __restrict__ c,
                           float* __restrict__ oh, float* __restrict__ oc) {
  int i = blockIdx.x * blockDim.x + threadIdx.x;
  if (i < BB * HH) { oh[i] = h[i]; oc[i] = c[i]; }
}

}  // namespace

extern "C" void kernel_launch(void* const* d_in, const int* in_sizes, int n_in,
                              void* d_out, int out_size, void* d_ws, size_t ws_size,
                              hipStream_t stream) {
  (void)in_sizes; (void)n_in; (void)out_size; (void)ws_size;

  const float* dec   = (const float*)d_in[0];
  const float* emb   = (const float*)d_in[1];
  const float* h0    = (const float*)d_in[2];
  const float* c0    = (const float*)d_in[3];
  const float* ctx   = (const float*)d_in[4];   // (L,B,H)
  const float* Wi    = (const float*)d_in[5];
  const float* bi    = (const float*)d_in[6];
  const float* Wh    = (const float*)d_in[7];
  const float* bh    = (const float*)d_in[8];
  const float* Wq_s  = (const float*)d_in[9];
  const float* bq_s  = (const float*)d_in[10];
  const float* Wr_s  = (const float*)d_in[11];
  const float* br_s  = (const float*)d_in[12];
  const float* v_s   = (const float*)d_in[13];
  const float* Wq_e  = (const float*)d_in[14];
  const float* bq_e  = (const float*)d_in[15];
  const float* Wr_e  = (const float*)d_in[16];
  const float* br_e  = (const float*)d_in[17];
  const float* v_e   = (const float*)d_in[18];
  const float* W_cls = (const float*)d_in[19];
  const float* b_cls = (const float*)d_in[20];

  float* out = (float*)d_out;
  // Output layout (flat f32, reference return order)
  const long o_probs_h = 0;
  const long o_sels_h  = o_probs_h + (long)TT * BB * LL;
  const long o_probs_t = o_sels_h  + (long)TT * BB;
  const long o_sels_t  = o_probs_t + (long)TT * BB * LL;
  const long o_cls     = o_sels_t  + (long)TT * BB;
  const long o_hx      = o_cls     + (long)TT * BB * 2;
  const long o_cx      = o_hx      + (long)BB * HH;

  // Workspace layout
  float* ws    = (float*)d_ws;
  float* e_s   = ws;                                  // B*H*L
  float* e_e   = e_s + (long)BB * HH * LL;            // B*H*L
  float* gates = e_e + (long)BB * HH * LL;            // B*4H
  float* hbuf  = gates + (long)BB * 4 * HH;           // B*H
  float* cbuf  = hbuf + (long)BB * HH;                // B*H
  float* xbuf  = cbuf + (long)BB * HH;                // B*E
  float* qbuf  = xbuf + (long)BB * EE;                // B*H
  float* gsb   = qbuf + (long)BB * HH;                // B*H
  float* geb   = gsb + (long)BB * HH;                 // B*H
  unsigned char* mh = (unsigned char*)(geb + (long)BB * HH);
  unsigned char* mt = mh + (long)BB * LL;
  int* idxh = (int*)(mt + (long)BB * LL);
  int* idxt = idxh + BB;

  // Init state + masks
  init_kernel<<<(BB * LL) / 256, 256, 0, stream>>>(dec, h0, c0, xbuf, hbuf, cbuf, mh, mt);

  // Precompute e_s/e_e[b,o,l] = sum_h Wr[o,h]*ctx[l,b,h] + br[o]
  // GEMM per b: A = Wr (o,k) lda=H; B^T rows = ctx + b*H, ldb = B*H; C ldc = L.
  {
    dim3 grid(LL / 128, HH / 16, BB);
    wmma_gemm_kernel<true><<<grid, 256, 0, stream>>>(
        Wr_s, HH, 0, ctx, BB * HH, HH, e_s, LL, (long)HH * LL, br_s, 1, 0, HH);
    wmma_gemm_kernel<true><<<grid, 256, 0, stream>>>(
        Wr_e, HH, 0, ctx, BB * HH, HH, e_e, LL, (long)HH * LL, br_e, 1, 0, HH);
  }

  for (int t = 0; t < TT; ++t) {
    // gates = x@Wi + bi + h@Wh + bh   (M=64, N=2048)
    {
      dim3 grid(4 * HH / 128, BB / 16, 1);
      wmma_gemm_kernel<false><<<grid, 256, 0, stream>>>(
          xbuf, EE, 0, Wi, 4 * HH, 0, gates, 4 * HH, 0, bi, 0, 0, EE);
      wmma_gemm_kernel<false><<<grid, 256, 0, stream>>>(
          hbuf, HH, 0, Wh, 4 * HH, 0, gates, 4 * HH, 0, bh, 0, 1, HH);
    }
    lstm_kernel<<<(BB * HH) / 256, 256, 0, stream>>>(gates, cbuf, hbuf);

    // q_s = h@Wq_s + bq_s  (M=64, N=512)
    {
      dim3 grid(HH / 128, BB / 16, 1);
      wmma_gemm_kernel<false><<<grid, 256, 0, stream>>>(
          hbuf, HH, 0, Wq_s, HH, 0, qbuf, HH, 0, bq_s, 0, 0, HH);
    }

    float* probs_s = out + o_probs_h + (long)t * BB * LL;
    attn_kernel<<<BB, LL, 0, stream>>>(e_s, qbuf, v_s, mh, probs_s,
                                       out + o_sels_h + (long)t * BB, idxh);
    wsum_kernel<<<dim3(BB, HH / 8), 256, 0, stream>>>(e_s, probs_s, gsb);

    // q_e = g_l_s@Wq_e + bq_e
    {
      dim3 grid(HH / 128, BB / 16, 1);
      wmma_gemm_kernel<false><<<grid, 256, 0, stream>>>(
          gsb, HH, 0, Wq_e, HH, 0, qbuf, HH, 0, bq_e, 0, 0, HH);
    }

    float* probs_e = out + o_probs_t + (long)t * BB * LL;
    attn_kernel<<<BB, LL, 0, stream>>>(e_e, qbuf, v_e, mt, probs_e,
                                       out + o_sels_t + (long)t * BB, idxt);
    // NOTE: reference uses e_s (not e_e) for g_l_e
    wsum_kernel<<<dim3(BB, HH / 8), 256, 0, stream>>>(e_s, probs_e, geb);

    cls_gather_kernel<<<BB, 256, 0, stream>>>(
        gsb, geb, W_cls, b_cls, emb, idxh, out + o_cls + (long)t * BB * 2, xbuf);
  }

  fin_kernel<<<(BB * HH) / 256, 256, 0, stream>>>(hbuf, cbuf, out + o_hx, out + o_cx);
}
